// BidafAttn_50182397887059
// MI455X (gfx1250) — compile-verified, matrix operand order checked by reading, not run on hardware
//
#include <hip/hip_runtime.h>

typedef __attribute__((ext_vector_type(16))) _Float16 v16h;
typedef __attribute__((ext_vector_type(4)))  _Float16 v4h;
typedef __attribute__((ext_vector_type(8)))  float    v8f;

constexpr int B_  = 64;
constexpr int CL  = 1024;
constexpr int QL  = 128;
constexpr int DD  = 512;
#define NEGINF (-1e30f)

// ---------------- Kernel 0: q -> f16 (row + transposed), qw2 = q@w2 + bias ----
__global__ __launch_bounds__(128) void k_prep_q(
    const float* __restrict__ q, const float* __restrict__ w,
    const float* __restrict__ bias,
    _Float16* __restrict__ qh, _Float16* __restrict__ qhT,
    float* __restrict__ qw2b)
{
  const int bj = blockIdx.x;          // b*QL + j
  const int b  = bj >> 7;
  const int j  = bj & (QL - 1);
  const int t  = threadIdx.x;         // 128 threads * 4 elems = 512
  const int d  = t * 4;

  const float* qrow = q + (size_t)bj * DD;
  float4 q4 = *(const float4*)(qrow + d);
  float4 w2 = *(const float4*)(w + DD + d);
  float acc = q4.x*w2.x + q4.y*w2.y + q4.z*w2.z + q4.w*w2.w;

  v4h hv = { (_Float16)q4.x, (_Float16)q4.y, (_Float16)q4.z, (_Float16)q4.w };
  *(v4h*)(qh + (size_t)bj * DD + d) = hv;

  _Float16* qt = qhT + (size_t)b * DD * QL;
  qt[(size_t)(d+0)*QL + j] = hv[0];
  qt[(size_t)(d+1)*QL + j] = hv[1];
  qt[(size_t)(d+2)*QL + j] = hv[2];
  qt[(size_t)(d+3)*QL + j] = hv[3];

  __shared__ float red[128];
  red[t] = acc; __syncthreads();
  for (int s = 64; s > 0; s >>= 1) {
    if (t < s) red[t] += red[t + s];
    __syncthreads();
  }
  if (t == 0) qw2b[bj] = red[0] + bias[0];
}

// ---------------- Kernel 1: sim (WMMA) -> softmax -> c2q (WMMA) -> out[0:1024] ----
__global__ __launch_bounds__(256) void k_sim_c2q(
    const int*   __restrict__ q_mask,
    const float* __restrict__ c, const float* __restrict__ w,
    const _Float16* __restrict__ qh, const _Float16* __restrict__ qhT,
    const float* __restrict__ qw2b,
    float* __restrict__ out, float* __restrict__ smax_ws)
{
  __shared__ _Float16 Ah[64][520];     // c-tile * w3 as f16, padded stride (bank-spread)
  __shared__ float    sim_s[64][132];
  __shared__ _Float16 dist_s[64][136];
  __shared__ float    redc[64][4];
  __shared__ float    qrow_s[128];
  __shared__ float    qmask_s[128];

  const int t   = threadIdx.x;
  const int blk = blockIdx.x;
  const int b   = blk >> 4;            // 16 i-tiles of 64 rows per batch
  const int it  = blk & 15;
  const int i0  = it * 64;

  if (t < 128) {
    qrow_s[t]  = qw2b[b * QL + t];                                  // q@w2 + bias
    qmask_s[t] = (1.0f - (float)q_mask[b * QL + t]) * NEGINF;
  }

  // ---- stage c-tile: Ah = (c * w3) as f16; cw1 partial dots ----
  {
    const int r   = t >> 2;            // 0..63
    const int sub = t & 3;             // 4 threads per row, 128 elems each
    const float* crow = c + (size_t)(b * CL + i0 + r) * DD + sub * 128;
    const float* w1p  = w + sub * 128;
    const float* w3p  = w + 2 * DD + sub * 128;
    float acc = 0.f;
#pragma unroll
    for (int kk = 0; kk < 128; kk += 4) {
      float4 c4  = *(const float4*)(crow + kk);
      float4 w14 = *(const float4*)(w1p + kk);
      float4 w34 = *(const float4*)(w3p + kk);
      acc += c4.x*w14.x + c4.y*w14.y + c4.z*w14.z + c4.w*w14.w;
      v4h hv = { (_Float16)(c4.x*w34.x), (_Float16)(c4.y*w34.y),
                 (_Float16)(c4.z*w34.z), (_Float16)(c4.w*w34.w) };
      *(v4h*)&Ah[r][sub * 128 + kk] = hv;
    }
    redc[r][sub] = acc;
  }
  __syncthreads();

  // ---- WMMA #1: sim[64][128] = Ah @ qh^T ----
  const int wv   = t >> 5;             // 8 waves
  const int lane = t & 31;
  const int i0w  = (wv & 3) * 16;      // wave's 16-row strip
  const int j0w  = (wv >> 2) * 64;     // wave's 64-col half
  const int lrow = lane & 15;
  const int lhi  = (lane >> 4) * 8;    // ISA 16-bit fragment K/M split

  v8f acc0 = {0,0,0,0,0,0,0,0};
  v8f acc[4] = {acc0, acc0, acc0, acc0};
  for (int k0 = 0; k0 < DD; k0 += 32) {
    v16h afrag;
    {
      const _Float16* ap = &Ah[i0w + lrow][k0 + lhi];
      ((uint4*)&afrag)[0] = *(const uint4*)ap;         // K 0..7 (+16..23)
      ((uint4*)&afrag)[1] = *(const uint4*)(ap + 16);
    }
#pragma unroll
    for (int jt = 0; jt < 4; ++jt) {
      const int jj = j0w + jt * 16 + lrow;
      const _Float16* bp = qh + ((size_t)b * QL + jj) * DD + k0 + lhi;
      v16h bfrag;
      ((uint4*)&bfrag)[0] = *(const uint4*)bp;
      ((uint4*)&bfrag)[1] = *(const uint4*)(bp + 16);
      acc[jt] = __builtin_amdgcn_wmma_f32_16x16x32_f16(
          false, afrag, false, bfrag, (short)0, acc[jt], false, false);
    }
  }
#pragma unroll
  for (int jt = 0; jt < 4; ++jt)
#pragma unroll
    for (int r = 0; r < 8; ++r)
      sim_s[i0w + r + lhi][j0w + jt * 16 + lrow] = acc[jt][r];
  __syncthreads();

  // ---- row softmax over j (threads 0..63, one row each) ----
  if (t < 64) {
    const float cw1 = redc[t][0] + redc[t][1] + redc[t][2] + redc[t][3];
    float mu = NEGINF, mm = NEGINF;
    for (int j = 0; j < QL; ++j) {
      float v = sim_s[t][j] + cw1 + qrow_s[j];
      sim_s[t][j] = v;
      mu = fmaxf(mu, v);                 // unmasked max -> s_max
      mm = fmaxf(mm, v + qmask_s[j]);    // masked max for softmax
    }
    smax_ws[b * CL + i0 + t] = mu;
    float sum = 0.f;
    for (int j = 0; j < QL; ++j) sum += __expf(sim_s[t][j] + qmask_s[j] - mm);
    const float inv = 1.0f / sum;
    for (int j = 0; j < QL; ++j)
      dist_s[t][j] = (_Float16)(__expf(sim_s[t][j] + qmask_s[j] - mm) * inv);
  }
  __syncthreads();

  // ---- WMMA #2: c2q[64][512] = dist @ q ; write out chunks 1 & 2 ----
  const int d0base = (wv >> 2) * 256;
  v16h afr[4];
#pragma unroll
  for (int jk = 0; jk < 4; ++jk) {
    const _Float16* ap = &dist_s[i0w + lrow][jk * 32 + lhi];
    ((uint4*)&afr[jk])[0] = *(const uint4*)ap;
    ((uint4*)&afr[jk])[1] = *(const uint4*)(ap + 16);
  }
  for (int nt = 0; nt < 16; ++nt) {
    const int d0 = d0base + nt * 16;
    v8f o = {0,0,0,0,0,0,0,0};
#pragma unroll
    for (int jk = 0; jk < 4; ++jk) {
      const _Float16* bp = qhT + ((size_t)b * DD + d0 + lrow) * QL + jk * 32 + lhi;
      v16h bfrag;
      ((uint4*)&bfrag)[0] = *(const uint4*)bp;
      ((uint4*)&bfrag)[1] = *(const uint4*)(bp + 16);
      o = __builtin_amdgcn_wmma_f32_16x16x32_f16(
          false, afr[jk], false, bfrag, (short)0, o, false, false);
    }
    const int dd = d0 + lrow;
#pragma unroll
    for (int r = 0; r < 8; ++r) {
      const int i = i0 + i0w + r + lhi;
      const float cv = c[(size_t)(b * CL + i) * DD + dd];
      const size_t ob = (size_t)(b * CL + i) * (3 * DD);
      out[ob + dd]      = o[r];          // c2q
      out[ob + DD + dd] = cv * o[r];     // c * c2q
    }
  }
}

// ---------------- Kernel 2: softmax(s_max) over i, c_dash = dist^T @ c ----
__global__ __launch_bounds__(256) void k_cdash(
    const float* __restrict__ c, const int* __restrict__ c_mask,
    const float* __restrict__ smax, float* __restrict__ cdash)
{
  __shared__ float wsh[1024];
  __shared__ float red[256];
  const int b = blockIdx.x, t = threadIdx.x;

  float lm = NEGINF;
  for (int i = t; i < CL; i += 256) {
    float v = smax[b * CL + i] + (1.0f - (float)c_mask[b * CL + i]) * NEGINF;
    wsh[i] = v;
    lm = fmaxf(lm, v);
  }
  red[t] = lm; __syncthreads();
  for (int s = 128; s > 0; s >>= 1) { if (t < s) red[t] = fmaxf(red[t], red[t+s]); __syncthreads(); }
  const float gmax = red[0]; __syncthreads();

  float ls = 0.f;
  for (int i = t; i < CL; i += 256) { float e = __expf(wsh[i] - gmax); wsh[i] = e; ls += e; }
  red[t] = ls; __syncthreads();
  for (int s = 128; s > 0; s >>= 1) { if (t < s) red[t] += red[t+s]; __syncthreads(); }
  const float inv = 1.0f / red[0]; __syncthreads();

  float a0 = 0.f, a1 = 0.f;
  for (int i = 0; i < CL; ++i) {
    const float wv = wsh[i];
    const float* crow = c + (size_t)(b * CL + i) * DD;
    a0 += wv * crow[t];
    a1 += wv * crow[t + 256];
  }
  cdash[b * DD + t]       = a0 * inv;
  cdash[b * DD + t + 256] = a1 * inv;
}

// ---------------- Kernel 3: out chunk 3 = c * c_dash (broadcast over i) ----
__global__ __launch_bounds__(256) void k_out3(
    const float* __restrict__ c, const float* __restrict__ cdash,
    float* __restrict__ out)
{
  const size_t flat = (size_t)blockIdx.x * 256 + threadIdx.x;   // float4 index
  const int d4 = (int)(flat & 127);
  const int i  = (int)((flat >> 7) & 1023);
  const int b  = (int)(flat >> 17);
  float4 c4 = ((const float4*)c)[flat];
  float4 g4 = ((const float4*)cdash)[b * 128 + d4];
  float4 o;
  o.x = c4.x * g4.x; o.y = c4.y * g4.y; o.z = c4.z * g4.z; o.w = c4.w * g4.w;
  ((float4*)out)[(size_t)(b * CL + i) * 384 + 256 + d4] = o;
}

extern "C" void kernel_launch(void* const* d_in, const int* in_sizes, int n_in,
                              void* d_out, int out_size, void* d_ws, size_t ws_size,
                              hipStream_t stream) {
  const float* q      = (const float*)d_in[0];
  const int*   q_mask = (const int*)d_in[1];
  const float* c      = (const float*)d_in[2];
  const int*   c_mask = (const int*)d_in[3];
  const float* w      = (const float*)d_in[4];
  const float* bias   = (const float*)d_in[5];
  float* out = (float*)d_out;

  char* ws = (char*)d_ws;
  const size_t QH_BYTES = (size_t)B_ * QL * DD * sizeof(_Float16);  // 8 MiB
  _Float16* qh   = (_Float16*)(ws);
  _Float16* qhT  = (_Float16*)(ws + QH_BYTES);
  float*    qw2b = (float*)(ws + 2 * QH_BYTES);
  float*    smax = (float*)(ws + 2 * QH_BYTES + (size_t)B_ * QL * 4);
  float*    cdsh = (float*)(ws + 2 * QH_BYTES + (size_t)B_ * QL * 4 + (size_t)B_ * CL * 4);

  k_prep_q <<<B_ * QL,            128, 0, stream>>>(q, w, bias, qh, qhT, qw2b);
  k_sim_c2q<<<B_ * (CL / 64),     256, 0, stream>>>(q_mask, c, w, qh, qhT, qw2b, out, smax);
  k_cdash  <<<B_,                 256, 0, stream>>>(c, c_mask, smax, cdsh);
  k_out3   <<<(B_ * CL * DD / 4) / 256, 256, 0, stream>>>(c, cdsh, out);
}
